// LightGCN_10849087390119
// MI455X (gfx1250) — matrix operand order repaired and checked by reading, not compiled, not run on hardware
//
#include <hip/hip_runtime.h>
#include <hip/hip_bf16.h>

typedef __attribute__((ext_vector_type(16))) _Float16 v16h;
typedef __attribute__((ext_vector_type(8)))  float    v8f;

#define N_NODES 100000
#define N_EDGES 3200000
#define IN_DIM  256
#define EMB_DIM 64
#define BATCH   4096
#define M_TILES (N_NODES / 16)   // 6250

// ---------------------------------------------------------------------------
// K0: repack W [256][64] f32 into WMMA B-fragment layout, f16.
// Fragment (kk, nt) is a 32x16 (KxN) tile: K = kk*32.., N = nt*16..
// Lane L holds column n = nt*16 + (L%16); K-range base = (L<16 ? 0 : 16);
// half j (0..15) = W[kk*32 + kbase + j][n].  16 halfs = 32B per lane.
// ---------------------------------------------------------------------------
__global__ void prep_w_kernel(const float* __restrict__ W, _Float16* __restrict__ wsB) {
    int p = blockIdx.x * blockDim.x + threadIdx.x;    // (frag, lane) pair id
    if (p >= 32 * 32) return;                          // 8 k-steps * 4 n-tiles = 32 frags
    int frag = p >> 5;
    int L    = p & 31;
    int kk   = frag >> 2;
    int nt   = frag & 3;
    int kbase = kk * 32 + ((L >> 4) ? 16 : 0);
    int n     = nt * 16 + (L & 15);
    _Float16* dst = wsB + (size_t)p * 16;
    #pragma unroll
    for (int j = 0; j < 16; ++j)
        dst[j] = (_Float16)W[(size_t)(kbase + j) * EMB_DIM + n];
}

// ---------------------------------------------------------------------------
// K1: zero h (N_NODES*64 floats) plus the 2 scalar accumulators that follow it
// ---------------------------------------------------------------------------
__global__ void zero_kernel(float* __restrict__ p, int n) {
    int i = blockIdx.x * blockDim.x + threadIdx.x;
    if (i < n) p[i] = 0.0f;
}

// ---------------------------------------------------------------------------
// K2: xw = feats @ W via v_wmma_f32_16x16x32_f16.
// 1 wave per 16-row tile; 4 accumulators cover N=64; K loop 8 x 32.
// A fragment (16x32 f16): lane L -> row m0 + L%16, K base (L/16)*16;
// halves 0..15 = K+0..15 (converted f32 -> f16).
// ---------------------------------------------------------------------------
__global__ void gemm_xw_kernel(const float* __restrict__ feats,
                               const _Float16* __restrict__ wsB,
                               float* __restrict__ xw) {
    int wave = threadIdx.x >> 5;
    int lane = threadIdx.x & 31;
    int tile = blockIdx.x * 8 + wave;
    if (tile >= M_TILES) return;                 // wave-uniform: EXEC stays all-1s
    int m0     = tile * 16;
    int half16 = lane >> 4;                      // 0 or 1
    int l16    = lane & 15;
    const float* arowp = feats + (size_t)(m0 + l16) * IN_DIM + half16 * 16;
    const v16h* bfr = (const v16h*)wsB;          // 32B per lane per fragment

    v8f acc0 = {}, acc1 = {}, acc2 = {}, acc3 = {};
    #pragma unroll
    for (int kk = 0; kk < 8; ++kk) {
        const float4* pa = (const float4*)(arowp + kk * 32);
        float4 f0 = pa[0], f1 = pa[1];           // K  .. K+7
        float4 f2 = pa[2], f3 = pa[3];           // K+8 .. K+15
        float t[16] = { f0.x, f0.y, f0.z, f0.w,  f1.x, f1.y, f1.z, f1.w,
                        f2.x, f2.y, f2.z, f2.w,  f3.x, f3.y, f3.z, f3.w };
        v16h a;
        #pragma unroll
        for (int j = 0; j < 16; ++j) a[j] = (_Float16)t[j];

        v16h b0 = bfr[(kk * 4 + 0) * 32 + lane];
        v16h b1 = bfr[(kk * 4 + 1) * 32 + lane];
        v16h b2 = bfr[(kk * 4 + 2) * 32 + lane];
        v16h b3 = bfr[(kk * 4 + 3) * 32 + lane];

        acc0 = __builtin_amdgcn_wmma_f32_16x16x32_f16(false, a, false, b0, (short)0, acc0, false, false);
        acc1 = __builtin_amdgcn_wmma_f32_16x16x32_f16(false, a, false, b1, (short)0, acc1, false, false);
        acc2 = __builtin_amdgcn_wmma_f32_16x16x32_f16(false, a, false, b2, (short)0, acc2, false, false);
        acc3 = __builtin_amdgcn_wmma_f32_16x16x32_f16(false, a, false, b3, (short)0, acc3, false, false);
    }
    // C/D layout: VGPR i -> M = i + (lane>=16 ? 8 : 0), N = lane%16
    int rowoff = half16 * 8;
    #pragma unroll
    for (int i = 0; i < 8; ++i) {
        float* out = xw + (size_t)(m0 + i + rowoff) * EMB_DIM;
        out[ 0 + l16] = acc0[i];
        out[16 + l16] = acc1[i];
        out[32 + l16] = acc2[i];
        out[48 + l16] = acc3[i];
    }
}

// ---------------------------------------------------------------------------
// K3: h[row] += val * xw[col]  (COO SpMM, gathers hit L2: xw is 25.6 MB)
// 4 threads per edge, 16 dims each.
// ---------------------------------------------------------------------------
__global__ void edge_scatter_kernel(const int* __restrict__ erow,
                                    const int* __restrict__ ecol,
                                    const float* __restrict__ eval_,
                                    const float* __restrict__ xw,
                                    float* __restrict__ h) {
    long long t = (long long)blockIdx.x * blockDim.x + threadIdx.x;
    if (t >= (long long)N_EDGES * 4) return;
    int e = (int)(t >> 2);
    int q = (int)(t & 3);
    int r = erow[e];
    int c = ecol[e];
    float v = eval_[e];
    const float4* src = (const float4*)(xw + (size_t)c * EMB_DIM + q * 16);
    float* dst = h + (size_t)r * EMB_DIM + q * 16;
    #pragma unroll
    for (int i = 0; i < 4; ++i) {
        float4 x = src[i];
        atomicAdd(dst + i * 4 + 0, v * x.x);
        atomicAdd(dst + i * 4 + 1, v * x.y);
        atomicAdd(dst + i * 4 + 2, v * x.z);
        atomicAdd(dst + i * 4 + 3, v * x.w);
    }
}

// ---------------------------------------------------------------------------
// K4: emb = l2norm(tanh(h)) in-place; accumulate sum(emb*emb) into acc[0].
// One wave per row (2 dims/lane), wave32 shuffle reduction.
// ---------------------------------------------------------------------------
__global__ void norm_rows_kernel(float* __restrict__ h, float* __restrict__ acc_embsq) {
    __shared__ float lacc;
    if (threadIdx.x == 0) lacc = 0.0f;
    __syncthreads();
    int wave = threadIdx.x >> 5;
    int lane = threadIdx.x & 31;
    int row  = blockIdx.x * 8 + wave;
    if (row < N_NODES) {
        float* p = h + (size_t)row * EMB_DIM;
        float x0 = tanhf(p[lane]);
        float x1 = tanhf(p[lane + 32]);
        float ssq = x0 * x0 + x1 * x1;
        #pragma unroll
        for (int m = 16; m >= 1; m >>= 1) ssq += __shfl_xor(ssq, m, 32);
        float inv = rsqrtf(fmaxf(ssq, 1e-12f));
        p[lane]      = x0 * inv;
        p[lane + 32] = x1 * inv;
        if (lane == 0) atomicAdd(&lacc, ssq * inv * inv);
    }
    __syncthreads();
    if (threadIdx.x == 0) atomicAdd(acc_embsq, lacc);
}

// ---------------------------------------------------------------------------
// K5: BPR loss: sum_b softplus(-(o1.o2 - o1.on)) into acc[1]. Wave per sample.
// ---------------------------------------------------------------------------
__global__ void bpr_kernel(const float* __restrict__ emb,
                           const int* __restrict__ i1, const int* __restrict__ i2,
                           const int* __restrict__ ineg, float* __restrict__ acc_bpr) {
    __shared__ float lacc;
    if (threadIdx.x == 0) lacc = 0.0f;
    __syncthreads();
    int wave = threadIdx.x >> 5;
    int lane = threadIdx.x & 31;
    int b = blockIdx.x * 8 + wave;
    if (b < BATCH) {
        const float* p1 = emb + (size_t)i1[b]   * EMB_DIM;
        const float* p2 = emb + (size_t)i2[b]   * EMB_DIM;
        const float* pn = emb + (size_t)ineg[b] * EMB_DIM;
        float a0 = p1[lane], a1 = p1[lane + 32];
        float ui = a0 * p2[lane] + a1 * p2[lane + 32];
        float uj = a0 * pn[lane] + a1 * pn[lane + 32];
        #pragma unroll
        for (int m = 16; m >= 1; m >>= 1) {
            ui += __shfl_xor(ui, m, 32);
            uj += __shfl_xor(uj, m, 32);
        }
        if (lane == 0) {
            float z = uj - ui;                            // -(y_ui - y_uj)
            float l = fmaxf(z, 0.0f) + log1pf(expf(-fabsf(z)));  // stable softplus
            atomicAdd(&lacc, l);
        }
    }
    __syncthreads();
    if (threadIdx.x == 0) atomicAdd(acc_bpr, lacc);
}

// ---------------------------------------------------------------------------
// K6: loss = (bpr + 1e-4 * 0.5 * embsq) / BATCH
// ---------------------------------------------------------------------------
__global__ void finalize_kernel(const float* __restrict__ acc, float* __restrict__ out) {
    out[0] = (acc[1] + 1e-4f * 0.5f * acc[0]) / (float)BATCH;
}

extern "C" void kernel_launch(void* const* d_in, const int* in_sizes, int n_in,
                              void* d_out, int out_size, void* d_ws, size_t ws_size,
                              hipStream_t stream) {
    (void)in_sizes; (void)n_in; (void)out_size; (void)ws_size;
    const float* feats = (const float*)d_in[0];
    const float* W     = (const float*)d_in[1];
    const int*   erow  = (const int*)d_in[2];
    const int*   ecol  = (const int*)d_in[3];
    const float* evalv = (const float*)d_in[4];
    const int*   idx1  = (const int*)d_in[5];
    const int*   idx2  = (const int*)d_in[6];
    const int*   negi  = (const int*)d_in[7];

    // workspace layout
    const size_t B_BYTES  = 32 * 32 * 16 * sizeof(_Float16);        // 32 KB
    const size_t XW_BYTES = (size_t)N_NODES * EMB_DIM * sizeof(float); // 25.6 MB
    char* ws = (char*)d_ws;
    _Float16* wsB = (_Float16*)ws;
    float* xw  = (float*)(ws + B_BYTES);
    float* h   = (float*)(ws + B_BYTES + XW_BYTES);
    float* acc = (float*)(ws + B_BYTES + 2 * XW_BYTES);  // acc[0]=embsq, acc[1]=bpr

    prep_w_kernel<<<4, 256, 0, stream>>>(W, wsB);

    int nz = N_NODES * EMB_DIM + 2;                      // h + both accumulators
    zero_kernel<<<(nz + 255) / 256, 256, 0, stream>>>(h, nz);

    gemm_xw_kernel<<<(M_TILES + 7) / 8, 256, 0, stream>>>(feats, wsB, xw);

    long long ethreads = (long long)N_EDGES * 4;
    edge_scatter_kernel<<<(int)((ethreads + 255) / 256), 256, 0, stream>>>(
        erow, ecol, evalv, xw, h);

    norm_rows_kernel<<<(N_NODES + 7) / 8, 256, 0, stream>>>(h, acc);

    bpr_kernel<<<(BATCH + 7) / 8, 256, 0, stream>>>(h, idx1, idx2, negi, acc + 1);

    finalize_kernel<<<1, 1, 0, stream>>>(acc, (float*)d_out);
}
